// PropagationGNN_62251255989004
// MI455X (gfx1250) — matrix operand order repaired and checked by reading, not compiled, not run on hardware
//
#include <hip/hip_runtime.h>
#include <hip/hip_bf16.h>

// ---------------------------------------------------------------------------
// PropagationGNN on MI455X (gfx1250): f16 WMMA GEMMs with fused gather/concat,
// f32 accumulation, register-staged + LDS double-buffered software pipeline.
// ---------------------------------------------------------------------------

typedef _Float16 h16;
typedef _Float16 h16x8  __attribute__((ext_vector_type(8)));
typedef _Float16 h16x16 __attribute__((ext_vector_type(16)));
typedef float    f32x8  __attribute__((ext_vector_type(8)));

static constexpr int NN    = 50000;
static constexpr int NE    = 800000;
static constexpr int HID   = 128;
static constexpr int DEPTH = 6;

// LDS tile strides (halfs), padded to avoid bank conflicts
#define A_STRIDE 40   // 32 K-halfs + 8 pad
#define B_STRIDE 136  // 128 N-halfs + 8 pad

// Force a global (addrspace 1) 16-byte load: avoids FLAT path when the
// pointer provenance is too complex for address-space inference.
__device__ __forceinline__ h16x8 gload8(const h16* p)
{
    typedef __attribute__((address_space(1))) const h16x8* gp_t;
    return *(gp_t)(unsigned long long)p;
}

// ---------------------------------------------------------------------------
// Main WMMA GEMM: C[M x 128] = act( A[M x KV] @ B[KV x 128] + bias )
// A is assembled from KV/128 gatherable 128-wide f16 sources (fused concat).
// Block: 256 threads = 8 waves; block tile 128(M) x 128(N); wave tile 32x64.
// KV is a template parameter: all steps/chunks/sources resolve at compile
// time. Pipeline: global->regs (step s+2), regs->LDS (step s+1, alt buffer),
// LDS->WMMA (step s); one barrier per step.
// Tail rows are loaded from clamped (row 0) pointers unconditionally: WMMA
// row m depends only on A row m, so garbage tail rows never reach memory.
// ---------------------------------------------------------------------------
template<int KV>
__global__ __launch_bounds__(256)
void gemm_wmma_n128(const h16* __restrict__ Bw, const float* __restrict__ bias,
                    int M,
                    const h16* __restrict__ src0, const int* __restrict__ idx0,
                    const h16* __restrict__ src1, const int* __restrict__ idx1,
                    const h16* __restrict__ src2, const int* __restrict__ idx2,
                    h16* __restrict__ out, int relu)
{
    constexpr int NCH = KV / 128;   // number of 128-wide A sources
    constexpr int NS  = KV / 32;    // K steps

    __shared__ h16 ldsA[2][128 * A_STRIDE];
    __shared__ h16 ldsB[2][32 * B_STRIDE];

    const int tid = threadIdx.x;
    const int m0  = blockIdx.x * 128;

    // Fixed per-thread staging coordinates (same rows every step).
    const int ar0  = tid >> 2;       // A rows ar0 and ar0+64
    const int aseg = tid & 3;        // 8-half segment within a 32-half K step
    const int br0  = tid >> 4;       // B rows br0 and br0+16
    const int bseg = tid & 15;

    const bool av0 = (m0 + ar0) < M;
    const bool av1 = (m0 + ar0 + 64) < M;

    const h16* __restrict__ srcs[3] = { src0, src1, src2 };
    const int* __restrict__ idxs[3] = { idx0, idx1, idx2 };

    // Resolve gathered A-row base pointers once (registers). Tail rows clamp
    // to row 0 so unconditional loads stay in-bounds.
    const h16* ap[NCH][2];
    #pragma unroll
    for (int c = 0; c < NCH; ++c) {
        const int g0 = m0 + ar0, g1 = g0 + 64;
        const int r0 = av0 ? (idxs[c] ? idxs[c][g0] : g0) : 0;
        const int r1 = av1 ? (idxs[c] ? idxs[c][g1] : g1) : 0;
        ap[c][0] = srcs[c] + (size_t)r0 * HID + aseg * 8;
        ap[c][1] = srcs[c] + (size_t)r1 * HID + aseg * 8;
    }

    const int lane = tid & 31;
    const int w    = tid >> 5;
    const int wm   = w & 3;   // 4 row-groups of 32
    const int wn   = w >> 2;  // 2 col-groups of 64

    f32x8 acc[2][4] = {};

    // Register-staged tiles for the software pipeline.
    h16x8 sa0, sa1, sb0, sb1;
    auto loadStep = [&](int st) {   // st is compile-time under full unroll
        const int c  = st >> 2;            // source chunk
        const int ko = (st & 3) * 32;      // K offset within chunk
        sa0 = gload8(ap[c][0] + ko);
        sa1 = gload8(ap[c][1] + ko);
        const h16* bp = Bw + (size_t)(st * 32) * HID + bseg * 8;
        sb0 = gload8(bp + (size_t)br0 * HID);
        sb1 = gload8(bp + (size_t)(br0 + 16) * HID);
    };
    auto storeStep = [&](int buf) {
        *(h16x8*)&ldsA[buf][ar0 * A_STRIDE + aseg * 8]        = sa0;
        *(h16x8*)&ldsA[buf][(ar0 + 64) * A_STRIDE + aseg * 8] = sa1;
        *(h16x8*)&ldsB[buf][br0 * B_STRIDE + bseg * 8]        = sb0;
        *(h16x8*)&ldsB[buf][(br0 + 16) * B_STRIDE + bseg * 8] = sb1;
    };
    auto computeStep = [&](int buf) {
        // A fragments (16-bit A 16x32 layout: lanes<16 hold K{0-7,16-23}).
        h16x16 af[2];
        #pragma unroll
        for (int sub = 0; sub < 2; ++sub) {
            const int arow = wm * 32 + sub * 16 + (lane & 15);
            const int kb   = (lane >> 4) * 8;
            const h16* pa = &ldsA[buf][arow * A_STRIDE + kb];
            h16x8 lo = *(const h16x8*)pa;
            h16x8 hi = *(const h16x8*)(pa + 16);
            af[sub] = __builtin_shufflevector(lo, hi,
                        0,1,2,3,4,5,6,7,8,9,10,11,12,13,14,15);
        }
        // B fragments (lane l = N col l%16, K{0-7,16-23} for lanes<16) + WMMA.
        #pragma unroll
        for (int t = 0; t < 4; ++t) {
            const h16* pb = &ldsB[buf][((lane >> 4) * 8) * B_STRIDE + wn * 64 + t * 16 + (lane & 15)];
            h16x16 bf;
            #pragma unroll
            for (int s = 0; s < 16; ++s)
                bf[s] = pb[((s & 7) + ((s >> 3) << 4)) * B_STRIDE];
            #pragma unroll
            for (int sub = 0; sub < 2; ++sub) {
                acc[sub][t] = __builtin_amdgcn_wmma_f32_16x16x32_f16(
                    false, af[sub], false, bf, (short)0, acc[sub][t],
                    false, false);
            }
        }
    };

    // Pipeline prologue: buffer 0 filled, step-1 data staged in registers.
    loadStep(0);
    storeStep(0);
    if (NS > 1) loadStep(1);
    __syncthreads();

    #pragma unroll
    for (int st = 0; st < NS; ++st) {
        if (st + 1 < NS) storeStep((st + 1) & 1);   // drain regs (step st+1)
        if (st + 2 < NS) loadStep(st + 2);          // prefetch step st+2
        computeStep(st & 1);
        if (st + 1 < NS) __syncthreads();           // one barrier per step
    }

    // ---- epilogue: bias + optional ReLU, store f16 ----
    #pragma unroll
    for (int sub = 0; sub < 2; ++sub) {
        #pragma unroll
        for (int t = 0; t < 4; ++t) {
            #pragma unroll
            for (int r = 0; r < 8; ++r) {
                const int row = m0 + wm * 32 + sub * 16 + r + ((lane >> 4) << 3);
                const int col = wn * 64 + t * 16 + (lane & 15);
                if (row < M) {
                    float v = acc[sub][t][r] + bias[col];
                    if (relu) v = v > 0.f ? v : 0.f;
                    out[(size_t)row * HID + col] = (h16)v;
                }
            }
        }
    }
}

// ---------------------------------------------------------------------------
// Tiny-K first layers (K = 2 or 3), f32 in -> ReLU -> f16 out
// ---------------------------------------------------------------------------
__global__ void smallk_l1(const float* __restrict__ in, int K,
                          const float* __restrict__ W, const float* __restrict__ b,
                          h16* __restrict__ out, int total)
{
    const int g = blockIdx.x * blockDim.x + threadIdx.x;
    if (g >= total) return;
    const int i = g >> 7, j = g & 127;
    const float* x = in + (size_t)i * K;
    float acc = b[j];
    for (int k = 0; k < K; ++k) acc += x[k] * W[k * HID + j];
    out[g] = (h16)(acc > 0.f ? acc : 0.f);
}

// FiLM: gb[256] = scalar(3) @ Wf(3x256) + bf
__global__ void film_vec(const float* __restrict__ s, const float* __restrict__ Wf,
                         const float* __restrict__ bf, float* __restrict__ gb)
{
    const int j = threadIdx.x;
    if (j < 256) {
        float acc = bf[j];
        for (int k = 0; k < 3; ++k) acc += s[k] * Wf[k * 256 + j];
        gb[j] = acc;
    }
}

__global__ void film_apply(h16* __restrict__ h, const float* __restrict__ gb, int total)
{
    const int g = blockIdx.x * blockDim.x + threadIdx.x;
    if (g >= total) return;
    const int j = g & 127;
    h[g] = (h16)(gb[j] * (float)h[g] + gb[128 + j]);
}

__global__ void zero_f32(float* __restrict__ p, int n)
{
    const int g = blockIdx.x * blockDim.x + threadIdx.x;
    if (g < n) p[g] = 0.f;
}

__global__ void count_deg(const int* __restrict__ col, float* __restrict__ cnt, int ne)
{
    const int g = blockIdx.x * blockDim.x + threadIdx.x;
    if (g < ne) atomicAdd(&cnt[col[g]], 1.0f);
}

__global__ void scatter_add(const h16* __restrict__ e, const int* __restrict__ col,
                            float* __restrict__ agg, int total)
{
    const int g = blockIdx.x * blockDim.x + threadIdx.x;
    if (g >= total) return;
    const int ei = g >> 7, j = g & 127;
    atomicAdd(&agg[(size_t)col[ei] * HID + j], (float)e[g]);
}

__global__ void agg_fin(const float* __restrict__ agg, const float* __restrict__ cnt,
                        h16* __restrict__ aggH, int total)
{
    const int g = blockIdx.x * blockDim.x + threadIdx.x;
    if (g >= total) return;
    float c = cnt[g >> 7];
    c = c < 1.f ? 1.f : c;
    aggH[g] = (h16)(agg[g] / c);
}

// node_dec layer 3: [N,128] @ [128,1] + b -> f32 out[N]
__global__ void dec_dot(const h16* __restrict__ h, const float* __restrict__ W,
                        const float* __restrict__ b, float* __restrict__ out, int n)
{
    const int i = blockIdx.x * blockDim.x + threadIdx.x;
    if (i >= n) return;
    const h16* p = h + (size_t)i * HID;
    float acc = b[0];
    for (int k = 0; k < HID; ++k) acc += (float)p[k] * W[k];
    out[i] = acc;
}

__global__ void cvt_f32_f16(const float* __restrict__ in, h16* __restrict__ out, int n)
{
    const int g = blockIdx.x * blockDim.x + threadIdx.x;
    if (g < n) out[g] = (h16)in[g];
}

// ---------------------------------------------------------------------------
// Host orchestration
// ---------------------------------------------------------------------------
extern "C" void kernel_launch(void* const* d_in, const int* in_sizes, int n_in,
                              void* d_out, int out_size, void* d_ws, size_t ws_size,
                              hipStream_t stream)
{
    (void)in_sizes; (void)n_in; (void)out_size; (void)ws_size;

    const float* x         = (const float*)d_in[0];
    const float* edge_attr = (const float*)d_in[1];
    const float* scalarv   = (const float*)d_in[2];
    const int*   eindex    = (const int*)d_in[3];   // harness ABI: integer -> int32
    const int*   rowIdx    = eindex;                // edge_index[0]
    const int*   colIdx    = eindex + NE;           // edge_index[1]

    // params flattened in JAX pytree (sorted-key) order:
    // blocks[0..5]{edge[3x(W,b)], node[3x(W,b)]}, edge_enc, film, node_dec, node_enc
    int p = 4;
    const float *beW[DEPTH][3], *beB[DEPTH][3], *bnW[DEPTH][3], *bnB[DEPTH][3];
    for (int d = 0; d < DEPTH; ++d) {
        for (int l = 0; l < 3; ++l) { beW[d][l] = (const float*)d_in[p++]; beB[d][l] = (const float*)d_in[p++]; }
        for (int l = 0; l < 3; ++l) { bnW[d][l] = (const float*)d_in[p++]; bnB[d][l] = (const float*)d_in[p++]; }
    }
    const float *eeW[3], *eeB[3];
    for (int l = 0; l < 3; ++l) { eeW[l] = (const float*)d_in[p++]; eeB[l] = (const float*)d_in[p++]; }
    const float* filmW = (const float*)d_in[p++];
    const float* filmB = (const float*)d_in[p++];
    const float *dcW[3], *dcB[3];
    for (int l = 0; l < 3; ++l) { dcW[l] = (const float*)d_in[p++]; dcB[l] = (const float*)d_in[p++]; }
    const float *neW[3], *neB[3];
    for (int l = 0; l < 3; ++l) { neW[l] = (const float*)d_in[p++]; neB[l] = (const float*)d_in[p++]; }

    // ---- workspace carve ----
    char* base = (char*)d_ws;
    size_t off = 0;
    auto carve = [&](size_t bytes) -> void* {
        void* r = base + off;
        off += (bytes + 255) & ~(size_t)255;
        return r;
    };
    h16*   hP     = (h16*)carve((size_t)NN * HID * 2);
    h16*   hQ     = (h16*)carve((size_t)NN * HID * 2);
    h16*   eP     = (h16*)carve((size_t)NE * HID * 2);
    h16*   eQ     = (h16*)carve((size_t)NE * HID * 2);
    float* aggF   = (float*)carve((size_t)NN * HID * 4);
    h16*   aggH   = (h16*)carve((size_t)NN * HID * 2);
    float* counts = (float*)carve((size_t)NN * 4);
    float* gb     = (float*)carve(256 * 4);

    auto cvt = [&](const float* src, int n) -> h16* {
        h16* dst = (h16*)carve((size_t)n * 2);
        cvt_f32_f16<<<(n + 255) / 256, 256, 0, stream>>>(src, dst, n);
        return dst;
    };

    // f16 copies of all WMMA-consumed weights
    h16 *beWh[DEPTH][3], *bnWh[DEPTH][3];
    for (int d = 0; d < DEPTH; ++d) {
        beWh[d][0] = cvt(beW[d][0], 384 * 128);
        beWh[d][1] = cvt(beW[d][1], 128 * 128);
        beWh[d][2] = cvt(beW[d][2], 128 * 128);
        bnWh[d][0] = cvt(bnW[d][0], 256 * 128);
        bnWh[d][1] = cvt(bnW[d][1], 128 * 128);
        bnWh[d][2] = cvt(bnW[d][2], 128 * 128);
    }
    h16* eeWh1 = cvt(eeW[1], 128 * 128);
    h16* eeWh2 = cvt(eeW[2], 128 * 128);
    h16* neWh1 = cvt(neW[1], 128 * 128);
    h16* neWh2 = cvt(neW[2], 128 * 128);
    h16* dcWh0 = cvt(dcW[0], 128 * 128);
    h16* dcWh1 = cvt(dcW[1], 128 * 128);

    auto gemm = [&](const h16* Bw, const float* bias, int M, int K,
                    const h16* s0, const int* i0, const h16* s1, const int* i1,
                    const h16* s2, const int* i2, h16* outp, int relu) {
        const int grid = (M + 127) / 128;
        if (K == 384) {
            gemm_wmma_n128<384><<<grid, 256, 0, stream>>>(
                Bw, bias, M, s0, i0, s1, i1, s2, i2, outp, relu);
        } else if (K == 256) {
            gemm_wmma_n128<256><<<grid, 256, 0, stream>>>(
                Bw, bias, M, s0, i0, s1, i1, s2, i2, outp, relu);
        } else {
            gemm_wmma_n128<128><<<grid, 256, 0, stream>>>(
                Bw, bias, M, s0, i0, s1, i1, s2, i2, outp, relu);
        }
    };

    const int nTot = NN * HID;  // 6.4M
    const int eTot = NE * HID;  // 102.4M

    // ---- in-degree counts (once) ----
    zero_f32<<<(NN + 255) / 256, 256, 0, stream>>>(counts, NN);
    count_deg<<<(NE + 255) / 256, 256, 0, stream>>>(colIdx, counts, NE);

    // ---- node encoder: 2 -> 128 -> 128 -> 128 ----
    smallk_l1<<<(nTot + 255) / 256, 256, 0, stream>>>(x, 2, neW[0], neB[0], hP, nTot);
    gemm(neWh1, neB[1], NN, 128, hP, nullptr, nullptr, nullptr, nullptr, nullptr, hQ, 1);
    gemm(neWh2, neB[2], NN, 128, hQ, nullptr, nullptr, nullptr, nullptr, nullptr, hP, 0);

    // ---- FiLM ----
    film_vec<<<1, 256, 0, stream>>>(scalarv, filmW, filmB, gb);
    film_apply<<<(nTot + 255) / 256, 256, 0, stream>>>(hP, gb, nTot);

    // ---- edge encoder: 3 -> 128 -> 128 -> 128 ----
    smallk_l1<<<(eTot + 255) / 256, 256, 0, stream>>>(edge_attr, 3, eeW[0], eeB[0], eP, eTot);
    gemm(eeWh1, eeB[1], NE, 128, eP, nullptr, nullptr, nullptr, nullptr, nullptr, eQ, 1);
    gemm(eeWh2, eeB[2], NE, 128, eQ, nullptr, nullptr, nullptr, nullptr, nullptr, eP, 0);

    h16 *hc = hP, *hs = hQ;   // current h / scratch
    h16 *ec = eP, *es = eQ;   // current e / scratch

    for (int d = 0; d < DEPTH; ++d) {
        // EdgeModel: MLP(concat(h[row], h[col], e)) — gather fused into GEMM
        gemm(beWh[d][0], beB[d][0], NE, 384, hc, rowIdx, hc, colIdx, ec, nullptr, es, 1);
        gemm(beWh[d][1], beB[d][1], NE, 128, es, nullptr, nullptr, nullptr, nullptr, nullptr, ec, 1);
        gemm(beWh[d][2], beB[d][2], NE, 128, ec, nullptr, nullptr, nullptr, nullptr, nullptr, es, 0);
        { h16* t = ec; ec = es; es = t; }   // new e in ec

        // scatter-mean over col
        zero_f32<<<(nTot + 255) / 256, 256, 0, stream>>>(aggF, nTot);
        scatter_add<<<(eTot + 255) / 256, 256, 0, stream>>>(ec, colIdx, aggF, eTot);
        agg_fin<<<(nTot + 255) / 256, 256, 0, stream>>>(aggF, counts, aggH, nTot);

        // NodeModel: MLP(concat(h, agg))
        gemm(bnWh[d][0], bnB[d][0], NN, 256, hc, nullptr, aggH, nullptr, nullptr, nullptr, hs, 1);
        gemm(bnWh[d][1], bnB[d][1], NN, 128, hs, nullptr, nullptr, nullptr, nullptr, nullptr, hc, 1);
        gemm(bnWh[d][2], bnB[d][2], NN, 128, hc, nullptr, nullptr, nullptr, nullptr, nullptr, hs, 0);
        { h16* t = hc; hc = hs; hs = t; }   // new h in hc
    }

    // ---- node decoder: 128 -> 128 -> 128 -> 1 ----
    gemm(dcWh0, dcB[0], NN, 128, hc, nullptr, nullptr, nullptr, nullptr, nullptr, hs, 1);
    gemm(dcWh1, dcB[1], NN, 128, hs, nullptr, nullptr, nullptr, nullptr, nullptr, hc, 1);
    dec_dot<<<(NN + 255) / 256, 256, 0, stream>>>(hc, dcW[2], dcB[2], (float*)d_out, NN);
}
